// VisionEncoderWrapper_9062380994924
// MI455X (gfx1250) — matrix operand order repaired and checked by reading, not compiled, not run on hardware
//
#include <hip/hip_runtime.h>
#include <hip/hip_fp16.h>
#include <math.h>

typedef _Float16 h16;
typedef __attribute__((ext_vector_type(16))) _Float16 v16h;
typedef __attribute__((ext_vector_type(8)))  float    v8f;

#define S_TOK 1600
#define CDIM  1536
#define HEADS 12
#define DHEAD 128
#define LAYERS 4
#define IDIM  4096
#define MDIM  6144
#define PDIM  1176
#define EPS_F 1e-6f

union Frag { v16h h; uint4 q[2]; };
union Stage16 { uint4 q[2]; h16 e[16]; };

static __device__ inline v8f vzero8() {
  v8f z = {0.f,0.f,0.f,0.f,0.f,0.f,0.f,0.f};
  return z;
}
// gelu_tanh(x) = 0.5x(1+tanh(c(x+0.044715x^3))) = x*sigmoid(2c(x+0.044715x^3))
static __device__ inline float gelu_fast(float x) {
  const float t = 1.5957691216057308f * (x + 0.044715f * x * x * x);
  return x / (1.f + __expf(-t));
}
static __device__ inline unsigned packh(h16 a, h16 b) {
  union { h16 h[2]; unsigned u; } x;
  x.h[0] = a; x.h[1] = b;
  return x.u;
}

// ---------------------------------------------------------------------------
// f16 GEMM: C[M,N](f32) = A[M,K](f16) * Bt[N,K](f16)^T (+bias)(+C)(+act)
// Block tile 256x128, BK=32, 8 waves in 4(M)x2(N) grid, wave tile 64x64
// -> 16 WMMA per 16 ds_load_b128. Double-buffered LDS, 1 barrier / K-step.
// Compile-time ACC (accumulate into C) and GELU (tanh-approx) epilogues.
// ---------------------------------------------------------------------------
template <bool ACC, bool GELU>
__global__ __launch_bounds__(256, 1) void gemm_wmma(
    const h16* __restrict__ A, const h16* __restrict__ Bt,
    float* __restrict__ C, const float* __restrict__ bias,
    int M, int N, int K)
{
  __shared__ h16 sA[2][256 * 40];
  __shared__ h16 sB[2][128 * 40];
  const int tid  = threadIdx.x;
  const int wid  = tid >> 5;
  const int lane = tid & 31;
  const int wm = (wid & 3) * 64;   // wave M offset in block tile
  const int wn = (wid >> 2) * 64;  // wave N offset in block tile
  const int bm = blockIdx.y * 256;
  const int bn = blockIdx.x * 128;

  v8f acc[4][4];
#pragma unroll
  for (int i = 0; i < 4; ++i)
#pragma unroll
    for (int j = 0; j < 4; ++j)
      acc[i][j] = vzero8();

  // global->register staging (A: row tid, 32 halves; B: row tid>>1, 16 halves)
  uint4 ra[4], rb[2];
  const int arow = tid;                 // 0..255
  const int brow = tid >> 1;            // 0..127
  const int bseg = (tid & 1) * 16;      // 0 / 16
  const int nk = (K + 31) >> 5;

  auto load_global = [&](int kt) {
    const int k0 = kt << 5;
    const int gr = bm + arow;
    const int gn = bn + brow;
#pragma unroll
    for (int g = 0; g < 4; ++g) {
      const int kk = k0 + g * 8;
      uint4 v = {0u, 0u, 0u, 0u};
      if (gr < M && (kk + 8) <= K) v = *(const uint4*)(A + (size_t)gr * K + kk);
      ra[g] = v;
    }
#pragma unroll
    for (int g = 0; g < 2; ++g) {
      const int kk = k0 + bseg + g * 8;
      uint4 v = {0u, 0u, 0u, 0u};
      if (gn < N && (kk + 8) <= K) v = *(const uint4*)(Bt + (size_t)gn * K + kk);
      rb[g] = v;
    }
  };
  auto store_lds = [&](int buf) {
#pragma unroll
    for (int g = 0; g < 4; ++g)
      *(uint4*)(sA[buf] + arow * 40 + g * 8) = ra[g];
#pragma unroll
    for (int g = 0; g < 2; ++g)
      *(uint4*)(sB[buf] + brow * 40 + bseg + g * 8) = rb[g];
  };

  load_global(0);
  store_lds(0);
  __syncthreads();

  const int fr  = lane & 15;
  const int kba = (lane >> 4) * 8;   // A frag: K {kba..+7, kba+16..+23}
  const int kbb = (lane >> 4) * 8;   // B frag: K {kbb..+7, kbb+16..+23}

  for (int kt = 0; kt < nk; ++kt) {
    const int cur = kt & 1, nxt = cur ^ 1;
    if (kt + 1 < nk) load_global(kt + 1);

    Frag af[4];
#pragma unroll
    for (int mt = 0; mt < 4; ++mt) {
      const h16* p = sA[cur] + (wm + mt * 16 + fr) * 40 + kba;
      af[mt].q[0] = *(const uint4*)(p);
      af[mt].q[1] = *(const uint4*)(p + 16);
    }
#pragma unroll
    for (int nt = 0; nt < 4; ++nt) {
      Frag bf;
      const h16* p = sB[cur] + (wn + nt * 16 + fr) * 40 + kbb;
      bf.q[0] = *(const uint4*)(p);
      bf.q[1] = *(const uint4*)(p + 16);
#pragma unroll
      for (int mt = 0; mt < 4; ++mt)
        acc[mt][nt] = __builtin_amdgcn_wmma_f32_16x16x32_f16(
            false, af[mt].h, false, bf.h, (short)0, acc[mt][nt], false, false);
    }

    if (kt + 1 < nk) store_lds(nxt);
    __syncthreads();
  }

  // Epilogue. N tiles are always in-range for this net (N % 128 == 0); only
  // the last M block tile is partial.
  float bb[4];
#pragma unroll
  for (int nt = 0; nt < 4; ++nt) {
    const int col = bn + wn + nt * 16 + (lane & 15);
    bb[nt] = (bias != nullptr) ? bias[col] : 0.f;
  }
  const int rbase = bm + wm + 8 * (lane >> 4);
  if (bm + 256 <= M) {
#pragma unroll
    for (int mt = 0; mt < 4; ++mt) {
#pragma unroll
      for (int nt = 0; nt < 4; ++nt) {
        const int col = bn + wn + nt * 16 + (lane & 15);
#pragma unroll
        for (int v = 0; v < 8; ++v) {
          const size_t idx = (size_t)(rbase + mt * 16 + v) * N + col;
          float val = acc[mt][nt][v] + bb[nt];
          if (GELU) val = gelu_fast(val);
          if (ACC)  val += C[idx];
          C[idx] = val;
        }
      }
    }
  } else {
#pragma unroll
    for (int mt = 0; mt < 4; ++mt) {
#pragma unroll
      for (int nt = 0; nt < 4; ++nt) {
        const int col = bn + wn + nt * 16 + (lane & 15);
#pragma unroll
        for (int v = 0; v < 8; ++v) {
          const int gr = rbase + mt * 16 + v;
          if (gr < M) {
            const size_t idx = (size_t)gr * N + col;
            float val = acc[mt][nt][v] + bb[nt];
            if (GELU) val = gelu_fast(val);
            if (ACC)  val += C[idx];
            C[idx] = val;
          }
        }
      }
    }
  }
}

// ---------------------------------------------------------------------------
// Flash attention, one head / 64 q-rows per workgroup (4 waves x 16 rows).
// qh/kh/vh are head-major [H][S][D] f16. Writes o back token-major [S][H*D].
// Double-buffered K/V tiles, 1 barrier per 32-key block.
// ---------------------------------------------------------------------------
__global__ __launch_bounds__(128, 1) void attn_flash(
    const h16* __restrict__ qh, const h16* __restrict__ kh,
    const h16* __restrict__ vh, h16* __restrict__ of)
{
  __shared__ h16 sK[2][32 * 128];    // [key][d]
  __shared__ h16 sVt[2][128 * 40];   // [d][key], padded stride
  __shared__ h16 sP[4][16 * 40];     // per-wave P scratch, padded stride

  const int tid  = threadIdx.x;
  const int wid  = tid >> 5;
  const int lane = tid & 31;
  const int hh   = blockIdx.y;
  const int qbase = blockIdx.x * 64 + wid * 16;

  const h16* Q  = qh + (size_t)hh * S_TOK * DHEAD;
  const h16* Kp = kh + (size_t)hh * S_TOK * DHEAD;
  const h16* Vp = vh + (size_t)hh * S_TOK * DHEAD;

  // Q fragments held in registers for the whole kernel (16 rows x 128 d).
  Frag qf[4];
  {
    const int qrow = qbase + (lane & 15);
    const int kba  = (lane >> 4) * 8;
#pragma unroll
    for (int kc = 0; kc < 4; ++kc) {
      const h16* p = Q + (size_t)qrow * DHEAD + kc * 32 + kba;
      qf[kc].q[0] = *(const uint4*)(p);
      qf[kc].q[1] = *(const uint4*)(p + 16);
    }
  }

  v8f o[8];
  float m[8], l[8];
#pragma unroll
  for (int i = 0; i < 8; ++i) { o[i] = vzero8(); m[i] = -1e30f; l[i] = 0.f; }

  const float scale = 0.08838834764831845f;  // 128^-0.5
  const int kr   = tid >> 2;  // 0..31 key row (K copy)
  const int seg  = tid & 3;   // 0..3  32-half segment
  const int pp   = tid >> 3;  // 0..15 key pair (V transpose)
  const int dseg = tid & 7;   // 0..7  16-d segment

  uint4 rk4[4];
  Stage16 rv0, rv1;
  auto load_kv = [&](int kblk) {
    const h16* src = Kp + (size_t)(kblk + kr) * DHEAD + seg * 32;
#pragma unroll
    for (int j = 0; j < 4; ++j) rk4[j] = *(const uint4*)(src + j * 8);
    const h16* v0 = Vp + (size_t)(kblk + 2 * pp) * DHEAD + dseg * 16;
    rv0.q[0] = *(const uint4*)(v0);
    rv0.q[1] = *(const uint4*)(v0 + 8);
    rv1.q[0] = *(const uint4*)(v0 + DHEAD);
    rv1.q[1] = *(const uint4*)(v0 + DHEAD + 8);
  };
  auto store_kv = [&](int buf) {
    h16* dst = sK[buf] + kr * 128 + seg * 32;
#pragma unroll
    for (int j = 0; j < 4; ++j) *(uint4*)(dst + j * 8) = rk4[j];
#pragma unroll
    for (int i = 0; i < 16; ++i) {
      const int d = dseg * 16 + i;
      *(unsigned*)(sVt[buf] + d * 40 + 2 * pp) = packh(rv0.e[i], rv1.e[i]);
    }
  };

  load_kv(0);
  store_kv(0);
  __syncthreads();

  const int fr  = lane & 15;
  const int kbb = (lane >> 4) * 8;
  const int half = lane >> 4;

  for (int kblk = 0; kblk < S_TOK; kblk += 32) {
    const int cur = (kblk >> 5) & 1, nxt = cur ^ 1;
    if (kblk + 32 < S_TOK) load_kv(kblk + 32);

    // S1 = Q @ K^T  (16 x 32)
    v8f sc[2];
    sc[0] = vzero8(); sc[1] = vzero8();
#pragma unroll
    for (int kc = 0; kc < 4; ++kc) {
      Frag kf[2];
#pragma unroll
      for (int nt = 0; nt < 2; ++nt) {
        const h16* p = sK[cur] + (nt * 16 + fr) * 128 + kc * 32 + kbb;
        kf[nt].q[0] = *(const uint4*)(p);
        kf[nt].q[1] = *(const uint4*)(p + 16);
      }
#pragma unroll
      for (int nt = 0; nt < 2; ++nt)
        sc[nt] = __builtin_amdgcn_wmma_f32_16x16x32_f16(
            false, qf[kc].h, false, kf[nt].h, (short)0, sc[nt], false, false);
    }

    // Online softmax, fully unrolled so all register indices are static.
    // Row r = v + 8*half; its 32 cols live on the 16 lanes of this half-wave
    // (xor masks 1,2,4,8 stay inside the half).
#pragma unroll
    for (int v = 0; v < 8; ++v) {
      float s0 = sc[0][v] * scale;
      float s1 = sc[1][v] * scale;
      float rm = fmaxf(s0, s1);
#pragma unroll
      for (int offm = 8; offm >= 1; offm >>= 1)
        rm = fmaxf(rm, __shfl_xor(rm, offm, 32));
      const float mnew = fmaxf(m[v], rm);
      const float corr = __expf(m[v] - mnew);
      const float p0 = __expf(s0 - mnew);
      const float p1 = __expf(s1 - mnew);
      float rs = p0 + p1;
#pragma unroll
      for (int offs = 8; offs >= 1; offs >>= 1)
        rs += __shfl_xor(rs, offs, 32);
      l[v] = l[v] * corr + rs;
      m[v] = mnew;
#pragma unroll
      for (int nt2 = 0; nt2 < 8; ++nt2) o[nt2][v] *= corr;
      const int prow = v + 8 * half;
      const int pc   = lane & 15;
      sP[wid][prow * 40 + pc]      = (h16)p0;
      sP[wid][prow * 40 + 16 + pc] = (h16)p1;
    }
    asm volatile("s_wait_dscnt 0" ::: "memory");

    // O += P @ V   (A = P 16x32, Bt = sVt [128 d][32 keys])
    Frag pf;
    {
      const h16* p = sP[wid] + (lane & 15) * 40 + (lane >> 4) * 8;
      pf.q[0] = *(const uint4*)(p);
      pf.q[1] = *(const uint4*)(p + 16);
    }
#pragma unroll
    for (int nt2 = 0; nt2 < 8; ++nt2) {
      Frag vf;
      const h16* p = sVt[cur] + (nt2 * 16 + fr) * 40 + kbb;
      vf.q[0] = *(const uint4*)(p);
      vf.q[1] = *(const uint4*)(p + 16);
      o[nt2] = __builtin_amdgcn_wmma_f32_16x16x32_f16(
          false, pf.h, false, vf.h, (short)0, o[nt2], false, false);
    }

    if (kblk + 32 < S_TOK) store_kv(nxt);
    __syncthreads();
  }

  // normalize + store token-major
#pragma unroll
  for (int nt2 = 0; nt2 < 8; ++nt2) {
    const int d = nt2 * 16 + (lane & 15);
#pragma unroll
    for (int v = 0; v < 8; ++v) {
      const int s = qbase + v + 8 * (lane >> 4);
      of[(size_t)s * CDIM + hh * DHEAD + d] = (h16)(o[nt2][v] / l[v]);
    }
  }
}

// ---------------------------------------------------------------------------
// Elementwise / staging kernels
// ---------------------------------------------------------------------------
__global__ void cast_f16(const float* __restrict__ src, h16* __restrict__ dst, int n) {
  int i = blockIdx.x * 256 + threadIdx.x;
  if (i < n) dst[i] = (h16)src[i];
}

// src [K][N] f32 -> dst [N][K] f16
__global__ void transpose_cast(const float* __restrict__ src, h16* __restrict__ dst,
                               int K, int N) {
  __shared__ float tile[32][33];
  const int bx = blockIdx.x * 32;  // N
  const int by = blockIdx.y * 32;  // K
  const int tx = threadIdx.x, ty = threadIdx.y;
#pragma unroll
  for (int i = 0; i < 4; ++i) {
    const int r = by + ty + i * 8;
    const int c = bx + tx;
    tile[ty + i * 8][tx] = (r < K && c < N) ? src[(size_t)r * N + c] : 0.f;
  }
  __syncthreads();
#pragma unroll
  for (int i = 0; i < 4; ++i) {
    const int nn = bx + ty + i * 8;
    const int kk = by + tx;
    if (nn < N && kk < K) dst[(size_t)nn * K + kk] = (h16)tile[tx][ty + i * 8];
  }
}

__global__ __launch_bounds__(256) void rmsnorm_cast(
    const float* __restrict__ x, const float* __restrict__ w,
    h16* __restrict__ out, int cols) {
  __shared__ float red[8];
  const int row = blockIdx.x;
  const int tid = threadIdx.x;
  const int wid = tid >> 5, lane = tid & 31;
  float ss = 0.f;
  for (int c = tid; c < cols; c += 256) {
    const float v = x[(size_t)row * cols + c];
    ss += v * v;
  }
#pragma unroll
  for (int off = 16; off >= 1; off >>= 1) ss += __shfl_xor(ss, off, 32);
  if (lane == 0) red[wid] = ss;
  __syncthreads();
  if (tid == 0) {
    float s = 0.f;
#pragma unroll
    for (int i = 0; i < 8; ++i) s += red[i];
    red[0] = rsqrtf(s / (float)cols + EPS_F);
  }
  __syncthreads();
  const float rs = red[0];
  for (int c = tid; c < cols; c += 256)
    out[(size_t)row * cols + c] = (h16)(x[(size_t)row * cols + c] * rs * w[c]);
}

// Per (token, head): q/k rmsnorm + RoPE, scatter q/k/v to head-major f16.
__global__ __launch_bounds__(128) void qkv_post(
    const float* __restrict__ qkv, const float* __restrict__ qw,
    const float* __restrict__ kw, const int* __restrict__ pos,
    h16* __restrict__ qh, h16* __restrict__ kh, h16* __restrict__ vh) {
  __shared__ float rq[4], rk[4];
  __shared__ float sqn[128], skn[128];
  const int s = blockIdx.x, hh = blockIdx.y;
  const int d = threadIdx.x;
  const int wid = d >> 5, lane = d & 31;
  const size_t base = (size_t)s * (3 * CDIM) + hh * DHEAD + d;
  const float q = qkv[base];
  const float k = qkv[base + CDIM];
  const float v = qkv[base + 2 * CDIM];
  float sq = q * q, sk = k * k;
#pragma unroll
  for (int off = 16; off >= 1; off >>= 1) {
    sq += __shfl_xor(sq, off, 32);
    sk += __shfl_xor(sk, off, 32);
  }
  if (lane == 0) { rq[wid] = sq; rk[wid] = sk; }
  __syncthreads();
  const float sumq = rq[0] + rq[1] + rq[2] + rq[3];
  const float sumk = rk[0] + rk[1] + rk[2] + rk[3];
  const float qn = q * rsqrtf(sumq / 128.f + EPS_F) * qw[d];
  const float kn = k * rsqrtf(sumk / 128.f + EPS_F) * kw[d];
  sqn[d] = qn; skn[d] = kn;
  __syncthreads();
  const float qr = (d < 64) ? -sqn[d + 64] : sqn[d - 64];
  const float kr = (d < 64) ? -skn[d + 64] : skn[d - 64];
  const int j = d & 63;
  const int axis = j >> 5;
  // inv_freq = 10000^(-(j%32)/32) = exp(-ln(1e4) * (j%32)/32)
  const float fr = __expf(-(float)(j & 31) * (9.21034037197618f / 32.f));
  const float ang = (float)pos[s * 2 + axis] * fr;
  float sn, cs;
  __sincosf(ang, &sn, &cs);
  const size_t ho = ((size_t)hh * S_TOK + s) * DHEAD + d;
  qh[ho] = (h16)(qn * cs + qr * sn);
  kh[ho] = (h16)(kn * cs + kr * sn);
  vh[ho] = (h16)v;
}

__global__ void silu_mul_cast(const float* __restrict__ g, const float* __restrict__ u,
                              h16* __restrict__ out, int n) {
  int i = blockIdx.x * 256 + threadIdx.x;
  if (i < n) {
    const float gv = g[i];
    out[i] = (h16)((gv / (1.f + __expf(-gv))) * u[i]);
  }
}

// A_merged[t][c*4 + i*2 + j] = hn[t*4 + i*2 + j][c]
__global__ void merge_gather(const h16* __restrict__ hn, h16* __restrict__ am, int n) {
  int i = blockIdx.x * 256 + threadIdx.x;
  if (i < n) {
    const int t = i / MDIM;
    const int r = i % MDIM;
    const int c = r >> 2;
    const int ij = r & 3;
    am[i] = hn[(size_t)(t * 4 + ij) * CDIM + c];
  }
}

// ---------------------------------------------------------------------------
extern "C" void kernel_launch(void* const* d_in, const int* in_sizes, int n_in,
                              void* d_out, int out_size, void* d_ws, size_t ws_size,
                              hipStream_t stream) {
  (void)in_sizes; (void)n_in; (void)out_size; (void)ws_size;
  const float* pixel_values = (const float*)d_in[0];
  const int*   pos_ids      = (const int*)d_in[1];
  const float* w_patch   = (const float*)d_in[3];
  const float* qkv_w     = (const float*)d_in[4];
  const float* qkv_b     = (const float*)d_in[5];
  const float* q_norm_w  = (const float*)d_in[6];
  const float* k_norm_w  = (const float*)d_in[7];
  const float* proj_w    = (const float*)d_in[8];
  const float* proj_b    = (const float*)d_in[9];
  const float* norm1_w   = (const float*)d_in[10];
  const float* norm2_w   = (const float*)d_in[11];
  const float* gate_w    = (const float*)d_in[12];
  const float* up_w      = (const float*)d_in[13];
  const float* down_w    = (const float*)d_in[14];
  const float* post_ln_w = (const float*)d_in[15];
  const float* conv_w    = (const float*)d_in[16];
  const float* conv_b    = (const float*)d_in[17];
  const float* fc1_w     = (const float*)d_in[18];
  const float* fc1_b     = (const float*)d_in[19];
  const float* fc2_w     = (const float*)d_in[20];
  const float* fc2_b     = (const float*)d_in[21];
  float* out = (float*)d_out;

  size_t off = 0;
  auto alloc = [&](size_t bytes) -> void* {
    off = (off + 255) & ~(size_t)255;
    void* p = (char*)d_ws + off;
    off += bytes;
    return p;
  };

  const int TOK2 = S_TOK / 4;  // 400 merged tokens
  float* h    = (float*)alloc((size_t)S_TOK * CDIM * 4);
  float* qkv  = (float*)alloc((size_t)S_TOK * 3 * CDIM * 4);
  h16*   xf   = (h16*)  alloc((size_t)S_TOK * CDIM * 2);
  h16*   of   = (h16*)  alloc((size_t)S_TOK * CDIM * 2);
  h16*   qh   = (h16*)  alloc((size_t)HEADS * S_TOK * DHEAD * 2);
  h16*   kh   = (h16*)  alloc((size_t)HEADS * S_TOK * DHEAD * 2);
  h16*   vh   = (h16*)  alloc((size_t)HEADS * S_TOK * DHEAD * 2);
  float* gbuf = (float*)alloc((size_t)S_TOK * IDIM * 4);
  float* ubuf = (float*)alloc((size_t)S_TOK * IDIM * 4);
  h16*   hm   = (h16*)  alloc((size_t)S_TOK * IDIM * 2);
  h16*   wt   = (h16*)  alloc((size_t)MDIM * CDIM * 2);   // weight staging (max C*M)
  h16*   pxf  = (h16*)  alloc((size_t)S_TOK * PDIM * 2);
  h16*   am   = (h16*)  alloc((size_t)TOK2 * MDIM * 2);
  float* dbuf = (float*)alloc((size_t)TOK2 * CDIM * 4);
  h16*   df   = (h16*)  alloc((size_t)TOK2 * CDIM * 2);
  float* ebuf = (float*)alloc((size_t)TOK2 * MDIM * 4);
  h16*   ef   = (h16*)  alloc((size_t)TOK2 * MDIM * 2);

  // flags: 0 = plain, 1 = accumulate, 2 = gelu
  auto gemm = [&](const h16* A, const h16* Bt, float* Cd, const float* bias,
                  int M, int N, int K, int flags) {
    dim3 g((N + 127) / 128, (M + 255) / 256);
    if (flags == 1)
      gemm_wmma<true,  false><<<g, dim3(256), 0, stream>>>(A, Bt, Cd, bias, M, N, K);
    else if (flags == 2)
      gemm_wmma<false, true ><<<g, dim3(256), 0, stream>>>(A, Bt, Cd, bias, M, N, K);
    else
      gemm_wmma<false, false><<<g, dim3(256), 0, stream>>>(A, Bt, Cd, bias, M, N, K);
  };
  auto tcast = [&](const float* src, h16* dst, int K, int N) {
    dim3 g((N + 31) / 32, (K + 31) / 32);
    transpose_cast<<<g, dim3(32, 8), 0, stream>>>(src, dst, K, N);
  };
  auto ecast = [&](const float* src, h16* dst, int n) {
    cast_f16<<<(n + 255) / 256, 256, 0, stream>>>(src, dst, n);
  };

  // ---- patch embedding: h = pixels @ w_patch ----
  ecast(pixel_values, pxf, S_TOK * PDIM);
  tcast(w_patch, wt, PDIM, CDIM);
  gemm(pxf, wt, h, nullptr, S_TOK, CDIM, PDIM, 0);

  // ---- transformer layers ----
  for (int l = 0; l < LAYERS; ++l) {
    // x = rmsnorm(h, n1)
    rmsnorm_cast<<<S_TOK, 256, 0, stream>>>(h, norm1_w + (size_t)l * CDIM, xf, CDIM);
    // qkv = x @ qkv_w + qkv_b
    tcast(qkv_w + (size_t)l * CDIM * 3 * CDIM, wt, CDIM, 3 * CDIM);
    gemm(xf, wt, qkv, qkv_b + (size_t)l * 3 * CDIM, S_TOK, 3 * CDIM, CDIM, 0);
    // q/k norm + rope, scatter to head-major
    qkv_post<<<dim3(S_TOK, HEADS), 128, 0, stream>>>(
        qkv, q_norm_w + (size_t)l * DHEAD, k_norm_w + (size_t)l * DHEAD,
        pos_ids, qh, kh, vh);
    // attention
    attn_flash<<<dim3(S_TOK / 64, HEADS), 128, 0, stream>>>(qh, kh, vh, of);
    // h += o @ proj_w + proj_b
    tcast(proj_w + (size_t)l * CDIM * CDIM, wt, CDIM, CDIM);
    gemm(of, wt, h, proj_b + (size_t)l * CDIM, S_TOK, CDIM, CDIM, 1);
    // x2 = rmsnorm(h, n2)
    rmsnorm_cast<<<S_TOK, 256, 0, stream>>>(h, norm2_w + (size_t)l * CDIM, xf, CDIM);
    // MLP: h += (silu(x2@gw) * (x2@uw)) @ dw
    tcast(gate_w + (size_t)l * CDIM * IDIM, wt, CDIM, IDIM);
    gemm(xf, wt, gbuf, nullptr, S_TOK, IDIM, CDIM, 0);
    tcast(up_w + (size_t)l * CDIM * IDIM, wt, CDIM, IDIM);
    gemm(xf, wt, ubuf, nullptr, S_TOK, IDIM, CDIM, 0);
    silu_mul_cast<<<(S_TOK * IDIM + 255) / 256, 256, 0, stream>>>(
        gbuf, ubuf, hm, S_TOK * IDIM);
    tcast(down_w + (size_t)l * IDIM * CDIM, wt, IDIM, CDIM);
    gemm(hm, wt, h, nullptr, S_TOK, CDIM, IDIM, 1);
  }

  // ---- post-LN + patch merger ----
  rmsnorm_cast<<<S_TOK, 256, 0, stream>>>(h, post_ln_w, xf, CDIM);
  merge_gather<<<(TOK2 * MDIM + 255) / 256, 256, 0, stream>>>(xf, am, TOK2 * MDIM);
  // conv_w flat [o][c*4+ij] is already N x K: plain cast, no transpose.
  ecast(conv_w, wt, CDIM * MDIM);
  gemm(am, wt, dbuf, conv_b, TOK2, CDIM, MDIM, 0);
  ecast(dbuf, df, TOK2 * CDIM);
  // fc1 + gelu
  tcast(fc1_w, wt, CDIM, MDIM);
  gemm(df, wt, ebuf, fc1_b, TOK2, MDIM, CDIM, 2);
  ecast(ebuf, ef, TOK2 * MDIM);
  // fc2 -> output
  tcast(fc2_w, wt, MDIM, CDIM);
  gemm(ef, wt, out, fc2_b, TOK2, CDIM, MDIM, 0);
}
